// TTT2TokenModel_25142738551249
// MI455X (gfx1250) — compile-verified, matrix operand order checked by reading, not run on hardware
//
#include <hip/hip_runtime.h>
#include <hip/hip_bf16.h>
#include <cmath>

// ---------------------------------------------------------------------------
// Types for CDNA5 WMMA (wave32)
// ---------------------------------------------------------------------------
typedef __bf16 bf16_t;
typedef __bf16 v16bf __attribute__((ext_vector_type(16)));
typedef float  v8f   __attribute__((ext_vector_type(8)));

union Frag {                 // 16x32 bf16 fragment = 8 VGPRs
    uint4          u[2];
    v16bf          v;
    unsigned short s[16];
};

static __device__ inline v8f vzero() {
    v8f z;
#pragma unroll
    for (int i = 0; i < 8; ++i) z[i] = 0.0f;
    return z;
}

static __device__ inline v8f wmma_bf16(const Frag& a, const Frag& b, v8f c) {
    return __builtin_amdgcn_wmma_f32_16x16x32_bf16(
        false, a.v, false, b.v, (short)0, c, false, false);
}

static __device__ inline float gelu_exact(float z) {
    return 0.5f * z * (1.0f + erff(z * 0.70710678118654752f));
}

// ---------------------------------------------------------------------------
// CDNA5 async global->LDS staging (ASYNCcnt path), with sync fallback
// ---------------------------------------------------------------------------
#if defined(__has_builtin)
#if __has_builtin(__builtin_amdgcn_global_load_async_to_lds_b128)
#define HAVE_ASYNC_LDS 1
#endif
#endif

#ifdef HAVE_ASYNC_LDS
typedef int i32x4_t __attribute__((ext_vector_type(4)));
typedef __attribute__((address_space(1))) i32x4_t as1_i32x4;
typedef __attribute__((address_space(3))) i32x4_t as3_i32x4;
static __device__ inline void async_b128(const bf16_t* g, bf16_t* l) {
    __builtin_amdgcn_global_load_async_to_lds_b128(
        (as1_i32x4*)g, (as3_i32x4*)l, 0, 0);
}
static __device__ inline void wait_async0() {
#if __has_builtin(__builtin_amdgcn_s_wait_asynccnt)
    __builtin_amdgcn_s_wait_asynccnt(0);
#else
    asm volatile("s_wait_asynccnt 0" ::: "memory");
#endif
}
static __device__ inline void wait_async3() {
#if __has_builtin(__builtin_amdgcn_s_wait_asynccnt)
    __builtin_amdgcn_s_wait_asynccnt(3);
#else
    asm volatile("s_wait_asynccnt 3" ::: "memory");
#endif
}
#else
static __device__ inline void async_b128(const bf16_t* g, bf16_t* l) {
    *(uint4*)l = *(const uint4*)g;     // sync fallback: load + ds_store
}
static __device__ inline void wait_async0() {}
static __device__ inline void wait_async3() {}
#endif

// ---------------------------------------------------------------------------
// Problem constants
// ---------------------------------------------------------------------------
#define BATCH 16
#define SEQ   1000
#define MROWS (BATCH * SEQ)      // 16000
#define EMB   512
#define NH    8
#define DHD   64
#define E3    1536
#define FFD   2048
#define CBN   4096
#define NLAY  4
#define SPAD  1024               // padded sequence for attention / VT

// ---------------------------------------------------------------------------
// fp32 -> bf16 conversion
// ---------------------------------------------------------------------------
__global__ void k_f32_to_bf16(const float* __restrict__ src,
                              bf16_t* __restrict__ dst, long n) {
    long i = (long)blockIdx.x * blockDim.x + threadIdx.x;
    if (i < n) dst[i] = (bf16_t)src[i];
}

// ---------------------------------------------------------------------------
// Embedding: x = codebook[tok] + pos_enc ; writes fp32 x0 and y (= residual)
// ---------------------------------------------------------------------------
__global__ void k_embed(const int* __restrict__ tok,
                        const float* __restrict__ cb,
                        const float* __restrict__ pe,
                        float* __restrict__ x0, float* __restrict__ y) {
    long i = (long)blockIdx.x * blockDim.x + threadIdx.x;
    if (i >= (long)MROWS * EMB) return;
    int  e = (int)(i & (EMB - 1));
    long row = i >> 9;                 // /512
    int  s = (int)(row % SEQ);
    int  t = tok[row];
    float v = cb[(long)t * EMB + e] + pe[(long)s * EMB + e];
    x0[i] = v; y[i] = v;
}

// ---------------------------------------------------------------------------
// LayerNorm over E=512 (one wave per row), optional GELU, bf16 output
// ---------------------------------------------------------------------------
__global__ __launch_bounds__(256) void k_layernorm(
    const float* __restrict__ x, const float* __restrict__ g,
    const float* __restrict__ b, bf16_t* __restrict__ out, int doGelu) {
    int lane = threadIdx.x & 31;
    long row = (long)blockIdx.x * 8 + (threadIdx.x >> 5);
    const float* xr = x + row * EMB;
    float xv[16];
    float s = 0.f;
#pragma unroll
    for (int i = 0; i < 16; ++i) { xv[i] = xr[lane + 32 * i]; s += xv[i]; }
#pragma unroll
    for (int off = 16; off >= 1; off >>= 1) s += __shfl_xor(s, off, 32);
    float mean = s * (1.0f / EMB);
    float vs = 0.f;
#pragma unroll
    for (int i = 0; i < 16; ++i) { float d = xv[i] - mean; vs += d * d; }
#pragma unroll
    for (int off = 16; off >= 1; off >>= 1) vs += __shfl_xor(vs, off, 32);
    float rstd = rsqrtf(vs * (1.0f / EMB) + 1e-5f);
    bf16_t* orow = out + row * EMB;
#pragma unroll
    for (int i = 0; i < 16; ++i) {
        int e = lane + 32 * i;
        float v = (xv[i] - mean) * rstd * g[e] + b[e];
        if (doGelu) v = gelu_exact(v);
        orow[e] = (bf16_t)v;
    }
}

// ---------------------------------------------------------------------------
// GEMM: out[M,N] = A[M,K](bf16) @ W[N,K]^T(bf16) (+bias)(+gelu)(+res + rs*val)
// 256 threads = 8 waves (4M x 2N); block tile 64x128; each wave 16x64.
// A (64x32) and B (128x32) k-slices double-buffered in LDS, staged with
// async global->LDS loads (ASYNCcnt) when available.
// M,N,K all tile-aligned here (no bounds checks).
// ---------------------------------------------------------------------------
__global__ __launch_bounds__(256) void k_gemm(
    const bf16_t* __restrict__ A, const bf16_t* __restrict__ W,
    const float* __restrict__ bias, const float* __restrict__ res,
    const float* __restrict__ resScalePtr,
    float* __restrict__ outF, bf16_t* __restrict__ outB,
    int N, int K, int doGelu) {
    __shared__ __attribute__((aligned(16))) bf16_t sA[2][64 * 32];
    __shared__ __attribute__((aligned(16))) bf16_t sB[2][128 * 32];

    int tid  = threadIdx.x;
    int lane = tid & 31;
    int wid  = tid >> 5;
    int r16  = lane & 15, hi = lane >> 4;
    int wm   = wid & 3;          // M sub-tile 0..3
    int wn   = wid >> 2;         // N group 0..1
    long tileM = (long)blockIdx.x * 64;
    int  tileN = blockIdx.y * 128;

    // staging addresses (per thread): 1 A chunk + 2 B chunks of 16B
    int aRowS = tid >> 2, aSub = tid & 3;
    const bf16_t* aG = A + (tileM + aRowS) * (long)K + aSub * 8;
    int bRow0 = tid >> 2,         bSub0 = tid & 3;
    int bRow1 = (tid + 256) >> 2, bSub1 = tid & 3;
    const bf16_t* bG0 = W + (long)(tileN + bRow0) * K + bSub0 * 8;
    const bf16_t* bG1 = W + (long)(tileN + bRow1) * K + bSub1 * 8;
    int aL  = aRowS * 32 + aSub * 8;
    int bL0 = bRow0 * 32 + bSub0 * 8;
    int bL1 = bRow1 * 32 + bSub1 * 8;

    v8f acc[4];
#pragma unroll
    for (int t = 0; t < 4; ++t) acc[t] = vzero();

    const int nk = K >> 5;
    // prologue: stage k-slice 0 into buffer 0
    async_b128(aG, &sA[0][aL]);
    async_b128(bG0, &sB[0][bL0]);
    async_b128(bG1, &sB[0][bL1]);

    for (int ks = 0; ks < nk; ++ks) {
        int cur = ks & 1;
        if (ks + 1 < nk) {
            int nxt = cur ^ 1, k0 = (ks + 1) << 5;
            async_b128(aG + k0, &sA[nxt][aL]);
            async_b128(bG0 + k0, &sB[nxt][bL0]);
            async_b128(bG1 + k0, &sB[nxt][bL1]);
            wait_async3();
        } else {
            wait_async0();
        }
        __syncthreads();

        Frag af;
        const bf16_t* ap = &sA[cur][(wm * 16 + r16) * 32 + hi * 8];
        af.u[0] = *(const uint4*)(ap);
        af.u[1] = *(const uint4*)(ap + 16);
#pragma unroll
        for (int t = 0; t < 4; ++t) {
            Frag bf;
            const bf16_t* bp = &sB[cur][(wn * 64 + t * 16 + r16) * 32 + hi * 8];
            bf.u[0] = *(const uint4*)(bp);
            bf.u[1] = *(const uint4*)(bp + 16);
            acc[t] = wmma_bf16(af, bf, acc[t]);
        }
        __syncthreads();
    }

    float rs = resScalePtr ? *resScalePtr : 1.0f;
#pragma unroll
    for (int t = 0; t < 4; ++t) {
        int col = tileN + wn * 64 + 16 * t + r16;
        float bv = bias ? bias[col] : 0.0f;
#pragma unroll
        for (int r = 0; r < 8; ++r) {
            long row = tileM + wm * 16 + r + 8 * hi;
            long idx = row * (long)N + col;
            float v = acc[t][r] + bv;
            if (doGelu) v = gelu_exact(v);
            if (res) v = res[idx] + rs * v;
            if (outF) outF[idx] = v;
            if (outB) outB[idx] = (bf16_t)v;
        }
    }
}

// ---------------------------------------------------------------------------
// V transpose: VT[b][h][d][kv] (kv padded to 1024, zero-filled) so attention
// P@V B-fragments are two contiguous b128 loads.
// ---------------------------------------------------------------------------
__global__ void k_transposeV(const bf16_t* __restrict__ qkv,
                             bf16_t* __restrict__ vt) {
    long i = (long)blockIdx.x * blockDim.x + threadIdx.x;  // B*H*64*1024
    if (i >= (long)BATCH * NH * DHD * SPAD) return;
    int kv = (int)(i & (SPAD - 1));
    int d  = (int)((i >> 10) & (DHD - 1));
    int h  = (int)((i >> 16) & (NH - 1));
    int b  = (int)(i >> 19);
    bf16_t v = (bf16_t)0.0f;
    if (kv < SEQ)
        v = qkv[((long)b * SEQ + kv) * E3 + 2 * EMB + h * DHD + d];
    vt[i] = v;
}

// ---------------------------------------------------------------------------
// Flash attention: one wave per (b, h, 16-query tile). S padded 1000 -> 1024.
// qkv layout: [B*S, 3E] bf16; q at col h*64+d, k at 512+...; V read from VT.
// ---------------------------------------------------------------------------
__global__ __launch_bounds__(32) void k_attention(
    const bf16_t* __restrict__ qkv, const bf16_t* __restrict__ vt,
    bf16_t* __restrict__ ctx) {
    int lane = threadIdx.x;
    int r16 = lane & 15, hi = lane >> 4;
    int bid = blockIdx.x;
    int qt = bid % 63;
    int h  = (bid / 63) % NH;
    int b  = bid / (63 * NH);
    int q0 = qt * 16;

    __shared__ __attribute__((aligned(16))) bf16_t P[16 * 32];

    int qrow = q0 + r16; if (qrow > SEQ - 1) qrow = SEQ - 1;
    const bf16_t* qb = qkv + ((long)b * SEQ + qrow) * E3 + h * DHD + hi * 8;
    Frag qa0, qa1;
    qa0.u[0] = *(const uint4*)(qb);       qa0.u[1] = *(const uint4*)(qb + 16);
    qa1.u[0] = *(const uint4*)(qb + 32);  qa1.u[1] = *(const uint4*)(qb + 48);

    const bf16_t* kb  = qkv + (long)b * SEQ * E3 + EMB + h * DHD + hi * 8;
    const bf16_t* vtb = vt + (((long)b * NH + h) << 16);   // [64][1024]

    float m[8], lsum[8];
    v8f o[4];
#pragma unroll
    for (int r = 0; r < 8; ++r) { m[r] = -3.0e38f; lsum[r] = 0.f; }
#pragma unroll
    for (int t = 0; t < 4; ++t) o[t] = vzero();

    for (int j0 = 0; j0 < SPAD; j0 += 32) {
        v8f sc[2];
#pragma unroll
        for (int sub = 0; sub < 2; ++sub) {
            int kv = j0 + 16 * sub + r16;
            int kvc = kv > SEQ - 1 ? SEQ - 1 : kv;
            const bf16_t* kp = kb + (long)kvc * E3;
            Frag k0f, k1f;
            k0f.u[0] = *(const uint4*)(kp);       k0f.u[1] = *(const uint4*)(kp + 16);
            k1f.u[0] = *(const uint4*)(kp + 32);  k1f.u[1] = *(const uint4*)(kp + 48);
            v8f c = vzero();
            c = wmma_bf16(qa0, k0f, c);
            c = wmma_bf16(qa1, k1f, c);
            bool valid = kv < SEQ;
#pragma unroll
            for (int r = 0; r < 8; ++r)
                sc[sub][r] = valid ? c[r] * 0.125f : -3.0e38f;
        }
        // online softmax (row stats live in 16-lane halves)
#pragma unroll
        for (int r = 0; r < 8; ++r) {
            float mx = fmaxf(sc[0][r], sc[1][r]);
#pragma unroll
            for (int off = 8; off >= 1; off >>= 1)
                mx = fmaxf(mx, __shfl_xor(mx, off, 32));
            float mn = fmaxf(m[r], mx);
            float alpha = __expf(m[r] - mn);
            float p0 = __expf(sc[0][r] - mn);
            float p1 = __expf(sc[1][r] - mn);
            sc[0][r] = p0; sc[1][r] = p1;
            float rsum = p0 + p1;
#pragma unroll
            for (int off = 8; off >= 1; off >>= 1)
                rsum += __shfl_xor(rsum, off, 32);
            lsum[r] = lsum[r] * alpha + rsum;
            m[r] = mn;
#pragma unroll
            for (int t = 0; t < 4; ++t) o[t][r] *= alpha;
        }
        // C-layout -> LDS -> A-fragment re-layout for P (16x32)
#pragma unroll
        for (int r = 0; r < 8; ++r) {
            P[(r + 8 * hi) * 32 + r16]      = (bf16_t)sc[0][r];
            P[(r + 8 * hi) * 32 + 16 + r16] = (bf16_t)sc[1][r];
        }
        __syncthreads();
        Frag pf;
        pf.u[0] = *(const uint4*)&P[r16 * 32 + hi * 8];
        pf.u[1] = *(const uint4*)&P[r16 * 32 + hi * 8 + 16];
        __syncthreads();
        // ctx += P(16x32) @ V(32x64); V fragments from transposed VT rows
#pragma unroll
        for (int t = 0; t < 4; ++t) {
            Frag vf;
            const bf16_t* vp = vtb + (long)(t * 16 + r16) * SPAD + j0 + hi * 8;
            vf.u[0] = *(const uint4*)(vp);
            vf.u[1] = *(const uint4*)(vp + 16);
            o[t] = wmma_bf16(pf, vf, o[t]);
        }
    }
    // finalize and store ctx[b, q, h*64 + d] as bf16
#pragma unroll
    for (int r = 0; r < 8; ++r) {
        int q = q0 + r + 8 * hi;
        if (q >= SEQ) continue;
        float inv = 1.0f / lsum[r];
        bf16_t* crow = ctx + ((long)b * SEQ + q) * EMB + h * DHD;
#pragma unroll
        for (int t = 0; t < 4; ++t)
            crow[t * 16 + r16] = (bf16_t)(o[t][r] * inv);
    }
}

// ---------------------------------------------------------------------------
// Codebook row norms ||c||^2 (one wave per row)
// ---------------------------------------------------------------------------
__global__ __launch_bounds__(256) void k_cbnorm(const float* __restrict__ cb,
                                                float* __restrict__ cn) {
    int lane = threadIdx.x & 31;
    long row = (long)blockIdx.x * 8 + (threadIdx.x >> 5);
    const float* r = cb + row * EMB;
    float s = 0.f;
#pragma unroll
    for (int i = 0; i < 16; ++i) { float v = r[lane + 32 * i]; s += v * v; }
#pragma unroll
    for (int off = 16; off >= 1; off >>= 1) s += __shfl_xor(s, off, 32);
    if (lane == 0) cn[row] = s;
}

// ---------------------------------------------------------------------------
// Nearest-neighbour: argmin_n ( ||c_n||^2 - 2 f.c_n )  (||f||^2 const per row)
// one wave per 16-row tile; loops 256 codebook tiles via WMMA.
// ---------------------------------------------------------------------------
__global__ __launch_bounds__(32) void k_argmin(
    const bf16_t* __restrict__ F, const bf16_t* __restrict__ CB,
    const float* __restrict__ cn, int* __restrict__ toks) {
    int lane = threadIdx.x;
    int r16 = lane & 15, hi = lane >> 4;
    long tileM = (long)blockIdx.x * 16;

    uint4 alo[16], ahi16[16];
    const bf16_t* ar = F + (tileM + r16) * (long)EMB + hi * 8;
#pragma unroll
    for (int ks = 0; ks < 16; ++ks) {
        alo[ks]   = *(const uint4*)(ar + ks * 32);
        ahi16[ks] = *(const uint4*)(ar + ks * 32 + 16);
    }
    float minv[8];
    int   mini[8];
#pragma unroll
    for (int r = 0; r < 8; ++r) { minv[r] = 3.0e38f; mini[r] = 0; }

    for (int nt = 0; nt < CBN / 16; ++nt) {
        v8f acc = vzero();
        const bf16_t* br = CB + (long)(nt * 16 + r16) * EMB + hi * 8;
#pragma unroll
        for (int ks = 0; ks < 16; ++ks) {
            Frag af, bf;
            af.u[0] = alo[ks]; af.u[1] = ahi16[ks];
            bf.u[0] = *(const uint4*)(br + ks * 32);
            bf.u[1] = *(const uint4*)(br + ks * 32 + 16);
            acc = wmma_bf16(af, bf, acc);
        }
        int col = nt * 16 + r16;
        float c = cn[col];
#pragma unroll
        for (int r = 0; r < 8; ++r) {
            float d = c - 2.0f * acc[r];
            if (d < minv[r]) { minv[r] = d; mini[r] = col; }
        }
    }
#pragma unroll
    for (int r = 0; r < 8; ++r) {
#pragma unroll
        for (int off = 8; off >= 1; off >>= 1) {
            float ov = __shfl_xor(minv[r], off, 32);
            int   oi = __shfl_xor(mini[r], off, 32);
            if (ov < minv[r] || (ov == minv[r] && oi < mini[r])) {
                minv[r] = ov; mini[r] = oi;
            }
        }
        if (r16 == 0) toks[tileM + r + 8 * hi] = mini[r];
    }
}

// ---------------------------------------------------------------------------
// Orchestration
// ---------------------------------------------------------------------------
static inline size_t alignUp(size_t x) { return (x + 255) & ~(size_t)255; }

extern "C" void kernel_launch(void* const* d_in, const int* in_sizes, int n_in,
                              void* d_out, int out_size, void* d_ws, size_t ws_size,
                              hipStream_t stream) {
    (void)in_sizes; (void)n_in; (void)out_size; (void)ws_size;
    const int*   tok    = (const int*)  d_in[0];
    const float* cb     = (const float*)d_in[1];
    const float* pe     = (const float*)d_in[2];
    const float* qkv_w  = (const float*)d_in[3];
    const float* qkv_b  = (const float*)d_in[4];
    const float* out_w  = (const float*)d_in[5];
    const float* out_b  = (const float*)d_in[6];
    const float* ln1_g  = (const float*)d_in[7];
    const float* ln1_b  = (const float*)d_in[8];
    const float* ln2_g  = (const float*)d_in[9];
    const float* ln2_b  = (const float*)d_in[10];
    const float* ff1_w  = (const float*)d_in[11];
    const float* ff1_b  = (const float*)d_in[12];
    const float* ff2_w  = (const float*)d_in[13];
    const float* ff2_b  = (const float*)d_in[14];
    const float* fin_g  = (const float*)d_in[15];
    const float* fin_b  = (const float*)d_in[16];
    const float* pp1_w  = (const float*)d_in[17];
    const float* pp1_b  = (const float*)d_in[18];
    const float* ppln_g = (const float*)d_in[19];
    const float* ppln_b = (const float*)d_in[20];
    const float* pp2_w  = (const float*)d_in[21];
    const float* pp2_b  = (const float*)d_in[22];
    const float* res_w  = (const float*)d_in[23];
    const float* fc1_w  = (const float*)d_in[24];
    const float* fc1_b  = (const float*)d_in[25];
    const float* fcln_g = (const float*)d_in[26];
    const float* fcln_b = (const float*)d_in[27];
    const float* fc2_w  = (const float*)d_in[28];
    const float* fc2_b  = (const float*)d_in[29];

    // workspace layout
    char* p = (char*)d_ws;
    size_t off = 0;
    float*  x0     = (float*) (p + off); off = alignUp(off + (size_t)MROWS * EMB * 4);
    float*  y      = (float*) (p + off); off = alignUp(off + (size_t)MROWS * EMB * 4);
    float*  t1     = (float*) (p + off); off = alignUp(off + (size_t)MROWS * EMB * 4);
    float*  t2     = (float*) (p + off); off = alignUp(off + (size_t)MROWS * EMB * 4);
    bf16_t* a_bf   = (bf16_t*)(p + off); off = alignUp(off + (size_t)MROWS * EMB * 2);
    bf16_t* g_bf   = (bf16_t*)(p + off); off = alignUp(off + (size_t)MROWS * FFD * 2);
    bf16_t* qkv_bf = (bf16_t*)(p + off); off = alignUp(off + (size_t)MROWS * E3 * 2);
    bf16_t* ctx_bf = (bf16_t*)(p + off); off = alignUp(off + (size_t)MROWS * EMB * 2);
    bf16_t* vt_bf  = (bf16_t*)(p + off); off = alignUp(off + (size_t)BATCH * NH * DHD * SPAD * 2);
    bf16_t* wbf    = (bf16_t*)(p + off); off = alignUp(off + (size_t)15728640 * 2);
    float*  cnorm  = (float*) (p + off); off = alignUp(off + (size_t)CBN * 4);

    // bf16 weight arena offsets (elements)
    const size_t oQKV = 0;
    const size_t oOUT = oQKV + (size_t)NLAY * E3 * EMB;
    const size_t oFF1 = oOUT + (size_t)NLAY * EMB * EMB;
    const size_t oFF2 = oFF1 + (size_t)NLAY * FFD * EMB;
    const size_t oPP1 = oFF2 + (size_t)NLAY * EMB * FFD;
    const size_t oPP2 = oPP1 + (size_t)EMB * EMB;
    const size_t oFC1 = oPP2 + (size_t)EMB * EMB;
    const size_t oFC2 = oFC1 + (size_t)EMB * EMB;
    const size_t oCB  = oFC2 + (size_t)EMB * EMB;

    auto conv = [&](const float* s, bf16_t* d, long n) {
        k_f32_to_bf16<<<(unsigned)((n + 255) / 256), 256, 0, stream>>>(s, d, n);
    };
    conv(qkv_w, wbf + oQKV, (long)NLAY * E3 * EMB);
    conv(out_w, wbf + oOUT, (long)NLAY * EMB * EMB);
    conv(ff1_w, wbf + oFF1, (long)NLAY * FFD * EMB);
    conv(ff2_w, wbf + oFF2, (long)NLAY * EMB * FFD);
    conv(pp1_w, wbf + oPP1, (long)EMB * EMB);
    conv(pp2_w, wbf + oPP2, (long)EMB * EMB);
    conv(fc1_w, wbf + oFC1, (long)EMB * EMB);
    conv(fc2_w, wbf + oFC2, (long)EMB * EMB);
    conv(cb,    wbf + oCB,  (long)CBN * EMB);

    k_cbnorm<<<CBN / 8, 256, 0, stream>>>(cb, cnorm);

    k_embed<<<(unsigned)(((long)MROWS * EMB + 255) / 256), 256, 0, stream>>>(
        tok, cb, pe, x0, y);

    auto gemm = [&](const bf16_t* A, const bf16_t* W, const float* bias,
                    const float* res, const float* rsP, float* oF, bf16_t* oB,
                    int N, int K, int doGelu) {
        dim3 g(MROWS / 64, N / 128, 1);
        k_gemm<<<g, 256, 0, stream>>>(A, W, bias, res, rsP, oF, oB, N, K, doGelu);
    };

    const unsigned vtBlocks = (unsigned)(((long)BATCH * NH * DHD * SPAD + 255) / 256);

    for (int l = 0; l < NLAY; ++l) {
        k_layernorm<<<MROWS / 8, 256, 0, stream>>>(y, ln1_g + l * EMB, ln1_b + l * EMB, a_bf, 0);
        gemm(a_bf, wbf + oQKV + (size_t)l * E3 * EMB, qkv_b + l * E3,
             nullptr, nullptr, nullptr, qkv_bf, E3, EMB, 0);
        k_transposeV<<<vtBlocks, 256, 0, stream>>>(qkv_bf, vt_bf);
        k_attention<<<BATCH * NH * 63, 32, 0, stream>>>(qkv_bf, vt_bf, ctx_bf);
        gemm(ctx_bf, wbf + oOUT + (size_t)l * EMB * EMB, out_b + l * EMB,
             y, nullptr, y, nullptr, EMB, EMB, 0);
        k_layernorm<<<MROWS / 8, 256, 0, stream>>>(y, ln2_g + l * EMB, ln2_b + l * EMB, a_bf, 0);
        gemm(a_bf, wbf + oFF1 + (size_t)l * FFD * EMB, ff1_b + l * FFD,
             nullptr, nullptr, nullptr, g_bf, FFD, EMB, 1);
        gemm(g_bf, wbf + oFF2 + (size_t)l * EMB * FFD, ff2_b + l * EMB,
             y, nullptr, y, nullptr, EMB, FFD, 0);
    }

    // final LN -> pp chain -> enhanced = x0 + res_w * pp
    k_layernorm<<<MROWS / 8, 256, 0, stream>>>(y, fin_g, fin_b, a_bf, 0);
    gemm(a_bf, wbf + oPP1, pp1_b, nullptr, nullptr, t1, nullptr, EMB, EMB, 0);
    k_layernorm<<<MROWS / 8, 256, 0, stream>>>(t1, ppln_g, ppln_b, a_bf, 1);
    gemm(a_bf, wbf + oPP2, pp2_b, x0, res_w, t1, ctx_bf, EMB, EMB, 0);

    // feature_to_codebook: fc2 output is the first model output (fp32)
    gemm(ctx_bf, wbf + oFC1, fc1_b, nullptr, nullptr, t2, nullptr, EMB, EMB, 0);
    k_layernorm<<<MROWS / 8, 256, 0, stream>>>(t2, fcln_g, fcln_b, a_bf, 1);
    gemm(a_bf, wbf + oFC2, fc2_b, nullptr, nullptr, (float*)d_out, g_bf, EMB, EMB, 0);

    // quantize: nearest codebook entry -> tokens (int32 tail of d_out)
    int* tok_out = (int*)((float*)d_out + (size_t)MROWS * EMB);
    k_argmin<<<MROWS / 16, 32, 0, stream>>>(g_bf, wbf + oCB, cnorm, tok_out);
}
